// gen_En_11347303596497
// MI455X (gfx1250) — compile-verified, matrix operand order checked by reading, not run on hardware
//
#include <hip/hip_runtime.h>

// Problem constants (from reference): MODES=6, RES=128, N=8192, EY_SIZE=1024,
// TOTAL_SIZE = (8192+8)*128 = 1,049,600. hs input is unused by the reference.
#define MODES    6
#define RES      128
#define NWGUIDE  8192
#define EY_SIZE  1024
#define TOTALSZ  ((NWGUIDE + 8) * RES)
#define GROUP    16                    // waveguides per workgroup
#define SPAN     (15 * RES + EY_SIZE)  // 2944 output floats touched per group
#define LSTRIDE  25                    // LDS row stride (coprime with 64 banks)

typedef __attribute__((ext_vector_type(16))) _Float16 v16h;
typedef __attribute__((ext_vector_type(8)))  float    v8f;

__global__ void en_zero_kernel(float* __restrict__ out, int n) {
    int idx = blockIdx.x * blockDim.x + threadIdx.x;
    int stride = gridDim.x * blockDim.x;
    for (int i = idx; i < n; i += stride) out[i] = 0.0f;
}

__global__ __launch_bounds__(256)
void en_wmma_kernel(const float* __restrict__ U, const float* __restrict__ neff,
                    const float* __restrict__ Ey, float* __restrict__ En) {
    // LDS accumulator, transposed layout: local offset o = a*128 + s  ->  acc[s*25 + a]
    __shared__ float acc[128 * LSTRIDE];   // 12.5 KB

    const int tid    = threadIdx.x;
    const int lane   = tid & 31;
    const int waveId = tid >> 5;          // 0..7 (wave32, 8 waves)
    const int half   = lane >> 4;         // 0 | 1
    const int col    = lane & 15;         // WMMA column (waveguide) / A row (feature)
    const int i0     = blockIdx.x * GROUP;

    for (int i = tid; i < 128 * LSTRIDE; i += 256) acc[i] = 0.0f;
    __syncthreads();

    // Per-lane mode weights for waveguide column i0+col:  w = (neff/(neff+1)) * U
    float w[MODES];
    {
        const int iw = i0 + col;
        #pragma unroll
        for (int m = 0; m < MODES; ++m) {
            float nf  = neff[iw * MODES + m];
            w[m] = (nf / (nf + 1.0f)) * U[iw * MODES + m];
        }
    }

    // Block-diagonal B chunks: chunk c covers waveguides 4c..4c+3, K-local = wgl*8 + ms.
    // 16-bit operand layout: lane half-index h -> K = (h&7) + 8*half + 16*(h>>3).
    v16h B[4];
    #pragma unroll
    for (int c = 0; c < 4; ++c) {
        v16h b;
        #pragma unroll
        for (int h = 0; h < 16; ++h) {
            int k   = (h & 7) + 8 * half + ((h >> 3) << 4);
            int wgl = k >> 3;
            int ms  = k & 7;
            float v = ((4 * c + wgl) == col && ms < MODES) ? w[ms] : 0.0f;
            b[h] = (_Float16)v;
        }
        B[c] = b;
    }

    // 64 feature tiles of 16; wave w owns tiles {w, w+8, ...}  ->  disjoint LDS slices.
    for (int t = waveId; t < 64; t += 8) {
        const int f0 = t * 16;
        v8f cacc = {};
        #pragma unroll
        for (int c = 0; c < 4; ++c) {
            v16h a;
            #pragma unroll
            for (int h = 0; h < 16; ++h) {
                int k   = (h & 7) + 8 * half + ((h >> 3) << 4);
                int wgl = k >> 3;
                int ms  = k & 7;
                float v = 0.0f;
                if (ms < MODES) {   // compile-time after unroll; avoids OOB read on pad slots
                    size_t i = (size_t)(i0 + 4 * c + wgl);
                    v = Ey[(i * MODES + ms) * EY_SIZE + (size_t)f0 + col];
                }
                a[h] = (_Float16)v;
            }
            // D[f, wg] += A(16x32 f16) x B(32x16 f16), f32 accumulate
            cacc = __builtin_amdgcn_wmma_f32_16x16x32_f16(
                false, a, false, B[c], (short)0, cacc, false, false);
        }
        // D layout: VGPR r, lane L -> M = r + 8*half (feature-in-tile), N = col (waveguide).
        // Local output offset o = col*128 + f ; store transposed: s = f&127, a = col + f>>7.
        #pragma unroll
        for (int r = 0; r < 8; ++r) {
            int f = f0 + r + 8 * half;
            int s = f & 127;
            int a = col + (f >> 7);
            acc[s * LSTRIDE + a] += cacc[r];   // race-free: waves own disjoint s-slices
        }
    }
    __syncthreads();

    // Flush group window to global with float atomics (seams overlap neighbor groups).
    const size_t base = (size_t)i0 * RES + RES / 2;
    for (int o = tid; o < SPAN; o += 256) {
        int a = o >> 7;
        int s = o & 127;
        atomicAdd(&En[base + o], acc[s * LSTRIDE + a]);
    }
}

extern "C" void kernel_launch(void* const* d_in, const int* in_sizes, int n_in,
                              void* d_out, int out_size, void* d_ws, size_t ws_size,
                              hipStream_t stream) {
    // setup_inputs order: hs (unused), U, neff, Ey ; output: En (float32, TOTALSZ)
    const float* U    = (const float*)d_in[1];
    const float* neff = (const float*)d_in[2];
    const float* Ey   = (const float*)d_in[3];
    float* En = (float*)d_out;

    en_zero_kernel<<<1024, 256, 0, stream>>>(En, out_size);
    en_wmma_kernel<<<NWGUIDE / GROUP, 256, 0, stream>>>(U, neff, Ey, En);
}
